// SpikingGraphConvolution_51092930953444
// MI455X (gfx1250) — matrix operand order repaired and checked by reading, not compiled, not run on hardware
//
#include <hip/hip_runtime.h>

typedef __attribute__((ext_vector_type(16))) _Float16 v16h;
typedef __attribute__((ext_vector_type(8)))  float    v8f;

union Frag16 { v16h v; uint4 u[2]; };

#define ADJ_SCALE      4096.0f
#define INV_ADJ_SCALE  (1.0f / 4096.0f)

// ---------------------------------------------------------------------------
// Kernel 0: WT[g][f] = (f16) W[f][g]   (128x128, trivial)
// ---------------------------------------------------------------------------
__global__ void wt_kernel(const float* __restrict__ W, _Float16* __restrict__ WT, int F) {
    int idx = blockIdx.x * blockDim.x + threadIdx.x;   // idx = g*F + f
    WT[idx] = (_Float16)W[(idx % F) * F + (idx / F)];
}

// ---------------------------------------------------------------------------
// Kernel 1: supportT[b][g][n] = (f16) (x[b] @ W)[n][g]
// grid = B * (N/16) blocks of 256 threads (8 waves, wave w owns g-tile w*16).
// ---------------------------------------------------------------------------
__global__ __launch_bounds__(256) void support_kernel(
        const float* __restrict__ x, const _Float16* __restrict__ WT,
        _Float16* __restrict__ supportT, int N) {
    int ntiles = N >> 4;
    int b  = blockIdx.x / ntiles;
    int nt = blockIdx.x % ntiles;
    const float* xb  = x + ((size_t)b * N + (size_t)nt * 16) * 128;
    _Float16*    sTb = supportT + (size_t)b * 128 * N;

    __shared__ _Float16 xl[16 * 128];          // 4 KB, f16 x-tile
    int tid = threadIdx.x;

    #pragma unroll
    for (int i = 0; i < 2; ++i) {
        int e = tid + i * 256;
        int r = e >> 5, c = (e & 31) << 2;
        float4 f = *(const float4*)(xb + (size_t)r * 128 + c);
        _Float16* d = &xl[r * 128 + c];
        d[0] = (_Float16)f.x; d[1] = (_Float16)f.y;
        d[2] = (_Float16)f.z; d[3] = (_Float16)f.w;
    }
    __syncthreads();

    int wave = tid >> 5, lane = tid & 31;
    int h = lane >> 4, n16 = lane & 15;
    int g = wave * 16 + n16;

    v8f acc = {};
    #pragma unroll
    for (int kc = 0; kc < 128; kc += 32) {
        Frag16 A;
        const uint4* Ap = (const uint4*)(&xl[n16 * 128 + kc]);
        A.u[0] = Ap[h];
        A.u[1] = Ap[2 + h];
        Frag16 B;
        const uint4* Bp = (const uint4*)(WT + g * 128 + kc);
        B.u[0] = Bp[h];
        B.u[1] = Bp[2 + h];
        acc = __builtin_amdgcn_wmma_f32_16x16x32_f16(false, A.v, false, B.v,
                                                     (short)0, acc, false, false);
    }

    union { _Float16 hs[8]; uint4 u; } pk;
    #pragma unroll
    for (int r = 0; r < 8; ++r) pk.hs[r] = (_Float16)acc[r];
    *(uint4*)(sTb + (size_t)g * N + nt * 16 + h * 8) = pk.u;
}

// ---------------------------------------------------------------------------
// Kernel 2 (fused, double-buffered): I = adj @ support + bias, then adaptive-LIF
// scan + mean-spike output, all in registers.
// grid = B * (N/32); block = 256 (8 waves). K-tile = 64, 2 LDS buffers,
// ONE barrier per K-tile; next tile's global loads issued before compute.
// ---------------------------------------------------------------------------
__global__ __launch_bounds__(256) void gemm_lif_kernel(
        const float* __restrict__ adj, const _Float16* __restrict__ supportT,
        const float* __restrict__ bias, const int* __restrict__ tsp,
        float* __restrict__ out, int N) {
    int mtiles = N >> 5;
    int b  = blockIdx.x / mtiles;
    int mt = blockIdx.x % mtiles;
    const float*    adjb = adj + ((size_t)b * N + (size_t)mt * 32) * N;
    const _Float16* sTb  = supportT + (size_t)b * 128 * N;

    __shared__ _Float16 al[2 * 32 * 64];       // 2 x 4 KB adj tiles (f16, scaled)

    int tid  = threadIdx.x;
    int wave = tid >> 5, lane = tid & 31;
    int h = lane >> 4, n16 = lane & 15;
    int g = wave * 16 + n16;
    const _Float16* Bbase = sTb + (size_t)g * N;

    // staging map: thread t -> row t>>3 (0..31), cols (t&7)*8 .. +7  (two float4)
    int lr = tid >> 3;
    int lc = (tid & 7) << 3;
    const float* aldp = adjb + (size_t)lr * N + lc;
    _Float16*    ldst = &al[lr * 64 + lc];

    // software pipeline: tile 0 already in flight before the loop
    float4 f0 = *(const float4*)(aldp);
    float4 f1 = *(const float4*)(aldp + 4);

    v8f acc[2] = {};
    int buf = 0;
    for (int k0 = 0; k0 < N; k0 += 64) {
        // convert + store current tile into LDS buffer `buf`
        _Float16* d = ldst + buf * 2048;
        d[0] = (_Float16)(f0.x * ADJ_SCALE); d[1] = (_Float16)(f0.y * ADJ_SCALE);
        d[2] = (_Float16)(f0.z * ADJ_SCALE); d[3] = (_Float16)(f0.w * ADJ_SCALE);
        d[4] = (_Float16)(f1.x * ADJ_SCALE); d[5] = (_Float16)(f1.y * ADJ_SCALE);
        d[6] = (_Float16)(f1.z * ADJ_SCALE); d[7] = (_Float16)(f1.w * ADJ_SCALE);
        __syncthreads();

        // issue next tile's HBM loads now; they overlap the WMMAs below
        if (k0 + 64 < N) {
            f0 = *(const float4*)(aldp + k0 + 64);
            f1 = *(const float4*)(aldp + k0 + 68);
            __builtin_prefetch(aldp + k0 + 128, 0, 0);   // one tile further -> L2
        }

        const _Float16* lbuf = &al[buf * 2048];
        #pragma unroll
        for (int ks = 0; ks < 64; ks += 32) {
            Frag16 B;
            const uint4* Bp = (const uint4*)(Bbase + k0 + ks);
            B.u[0] = Bp[h];
            B.u[1] = Bp[2 + h];

            Frag16 A0, A1;
            const uint4* Ap0 = (const uint4*)(lbuf + n16 * 64 + ks);
            A0.u[0] = Ap0[h];  A0.u[1] = Ap0[2 + h];
            const uint4* Ap1 = (const uint4*)(lbuf + (16 + n16) * 64 + ks);
            A1.u[0] = Ap1[h];  A1.u[1] = Ap1[2 + h];

            acc[0] = __builtin_amdgcn_wmma_f32_16x16x32_f16(false, A0.v, false, B.v,
                                                            (short)0, acc[0], false, false);
            acc[1] = __builtin_amdgcn_wmma_f32_16x16x32_f16(false, A1.v, false, B.v,
                                                            (short)0, acc[1], false, false);
        }
        buf ^= 1;
    }

    // ---- fused adaptive-LIF epilogue (forward spike = hard Heaviside) ----
    int   ts = tsp[0];
    float bg = bias[g];
    const float mem_decay = 1.0f - 1.0f / 20.0f;   // tau_mem = 20
    float* outb = out + ((size_t)b * N + (size_t)mt * 32) * 128 + g;

    #pragma unroll
    for (int mi = 0; mi < 2; ++mi) {
        #pragma unroll
        for (int r = 0; r < 8; ++r) {
            float I = acc[mi][r] * INV_ADJ_SCALE + bg;
            float v = 0.0f, th = 1.0f, sacc = 0.0f;
            for (int t = 0; t < ts; ++t) {
                v = v * mem_decay + I;
                float s = (v >= th) ? 1.0f : 0.0f;
                v *= (1.0f - s);
                th += (1.0f - th) * (1.0f / 60.0f) + 0.1f * s;
                sacc += s;
            }
            int row = mi * 16 + h * 8 + r;     // D layout: M = h*8 + r
            outb[(size_t)row * 128] = sacc / (float)ts;
        }
    }
}

// ---------------------------------------------------------------------------
extern "C" void kernel_launch(void* const* d_in, const int* in_sizes, int n_in,
                              void* d_out, int out_size, void* d_ws, size_t ws_size,
                              hipStream_t stream) {
    (void)n_in; (void)out_size; (void)ws_size;
    const float* x    = (const float*)d_in[0];
    const float* adj  = (const float*)d_in[1];
    const float* W    = (const float*)d_in[2];
    const float* bias = (const float*)d_in[3];
    const int*   ts   = (const int*)d_in[4];

    const int F = 128;                                   // FIN == FOUT == 128
    long long bn  = (long long)in_sizes[0] / F;          // B*N
    long long bnn = (long long)in_sizes[1];              // B*N*N
    int N = (int)(bnn / bn);
    int B = (int)(bn / N);

    _Float16* WT       = (_Float16*)d_ws;                             // 32 KB
    _Float16* supportT = (_Float16*)((char*)d_ws + 32 * 1024);        // B*128*N f16

    wt_kernel<<<(F * F) / 256, 256, 0, stream>>>(W, WT, F);
    support_kernel<<<B * (N / 16), 256, 0, stream>>>(x, WT, supportT, N);
    gemm_lif_kernel<<<B * (N / 32), 256, 0, stream>>>(adj, supportT, bias, ts,
                                                      (float*)d_out, N);
}
